// CausalSelfAttention_16939351016061
// MI455X (gfx1250) — matrix operand order repaired and checked by reading, not compiled, not run on hardware
//
#include <hip/hip_runtime.h>
#include <hip/hip_bf16.h>

#define D_MODEL 2048
#define NH 16
#define NKV 4
#define HD 128
#define SEQT 2048

typedef __attribute__((ext_vector_type(16))) __bf16 v16bf;
typedef __attribute__((ext_vector_type(8)))  __bf16 v8bf;
typedef __attribute__((ext_vector_type(8)))  float  v8f;

static __device__ __forceinline__ unsigned short f2bf(float f) {
    unsigned u = __float_as_uint(f);
    u += 0x7FFFu + ((u >> 16) & 1u);          // round-to-nearest-even
    return (unsigned short)(u >> 16);
}
static __device__ __forceinline__ float bf2f(unsigned short h) {
    return __uint_as_float(((unsigned)h) << 16);
}
static __device__ __forceinline__ v16bf combine16(v8bf lo, v8bf hi) {
    v16bf r;
#pragma unroll
    for (int i = 0; i < 8; ++i) { r[i] = lo[i]; r[i + 8] = hi[i]; }
    return r;
}

// ---- CDNA5 async global->LDS copy (ASYNCcnt-tracked DMA-style copy) ----
// LDS offset = low 32 bits of the generic shared pointer (LDS aperture maps
// addr[31:0] directly to LDS). One call copies 16 bytes per lane.
static __device__ __forceinline__ void async_ld16(void* lds_dst, const void* gsrc) {
    unsigned lo = (unsigned)(unsigned long long)(uintptr_t)lds_dst;
    unsigned long long ga = (unsigned long long)(uintptr_t)gsrc;
    asm volatile("global_load_async_to_lds_b128 %0, %1, off"
                 :: "v"(lo), "v"(ga) : "memory");
}
static __device__ __forceinline__ void async_wait0() {
    asm volatile("s_wait_asynccnt 0x0" ::: "memory");
}

// ---------------------------------------------------------------- fp32 -> bf16
__global__ void f32_to_bf16_kernel(const float* __restrict__ in,
                                   unsigned short* __restrict__ out, size_t n) {
    size_t i = (size_t)blockIdx.x * blockDim.x + threadIdx.x;
    size_t stride = (size_t)gridDim.x * blockDim.x;
    for (; i < n; i += stride) out[i] = f2bf(in[i]);
}

// ------------------------------------------------- RoPE + (B,T,HH,hd)->(B,HH,T,hd)
__global__ void rope_permute_kernel(const unsigned short* __restrict__ in,
                                    unsigned short* __restrict__ out,
                                    int HHheads, int T, int doRope) {
    int bh = blockIdx.x;
    int b = bh / HHheads, h = bh % HHheads;
    int t = blockIdx.y;
    int d = threadIdx.x;                       // 0..63 (pair d, d+64)
    size_t ii = ((size_t)((size_t)b * T + t) * HHheads + h) * HD + d;
    float x1 = bf2f(in[ii]), x2 = bf2f(in[ii + 64]);
    float o1 = x1, o2 = x2;
    if (doRope) {
        float f = __expf(-(float)d * (9.2103403719761836f / 64.0f)); // theta^(-2d/128)
        float ang = (float)t * f;
        float c = __cosf(ang), s = __sinf(ang);
        o1 = x1 * c - x2 * s;
        o2 = x2 * c + x1 * s;
    }
    size_t oi = ((size_t)((size_t)b * HHheads + h) * T + t) * HD + d;
    out[oi] = f2bf(o1);
    out[oi + 64] = f2bf(o2);
}

// ---------------------------------------------------------------- bf16 GEMM
// C[M,N] = A[M,K] * B[K,N]; A,B bf16 row-major, C fp32 or bf16 row-major.
// 256 threads (8 waves), 128x128 macro tile, K-step 32, double-buffered LDS
// with async global->LDS staging for A.
static __device__ __forceinline__ void stage_A(unsigned short (*ldsA)[32],
        const unsigned short* __restrict__ A, int m0, int K, int k0, int tid) {
    int row = tid >> 1, kc = (tid & 1) * 16;
    const unsigned short* src = A + (size_t)(m0 + row) * K + k0 + kc;
    async_ld16(&ldsA[row][kc],     src);
    async_ld16(&ldsA[row][kc + 8], src + 8);
}
static __device__ __forceinline__ void stage_Bt(unsigned short (*ldsBt)[32],
        const unsigned short* __restrict__ Bm, int n0, int N, int k0, int tid) {
    int kk = tid >> 3, nc = (tid & 7) * 16;
    const unsigned short* src = Bm + (size_t)(k0 + kk) * N + n0 + nc;
#pragma unroll
    for (int i = 0; i < 16; ++i) ldsBt[nc + i][kk] = src[i];   // transpose scatter
}

__global__ __launch_bounds__(256) void gemm_bf16_wmma(
        const unsigned short* __restrict__ A, const unsigned short* __restrict__ Bm,
        float* __restrict__ Cf, unsigned short* __restrict__ Cb,
        int M, int N, int K, int storef32) {
    __shared__ alignas(32) unsigned short ldsA[2][128][32];   // [buf][m][k]
    __shared__ alignas(32) unsigned short ldsBt[2][128][32];  // [buf][n][k]

    int tid = threadIdx.x;
    int w = tid >> 5, lane = tid & 31;
    int lh = lane & 15, half = lane >> 4;
    int m0 = blockIdx.y * 128, n0 = blockIdx.x * 128;
    int wm = (w & 3) * 32, wn = (w >> 2) * 64;   // wave tile: 32(M) x 64(N)

    v8f acc[2][4] = {};

    // prologue: stage first tile into buffer 0
    stage_A(ldsA[0], A, m0, K, 0, tid);
    stage_Bt(ldsBt[0], Bm, n0, N, 0, tid);
    async_wait0();
    __syncthreads();

    int buf = 0;
    for (int k0 = 0; k0 < K; k0 += 32) {
        int nk = k0 + 32;
        if (nk < K) {                         // issue async copies for next tile
            stage_A(ldsA[buf ^ 1], A, m0, K, nk, tid);
            stage_Bt(ldsBt[buf ^ 1], Bm, n0, N, nk, tid);
            if (nk + 32 < K)
                __builtin_prefetch(Bm + (size_t)(nk + 32 + (tid >> 3)) * N + n0, 0, 0);
        }

        // compute on current buffer
        v16bf af[2];
#pragma unroll
        for (int mt = 0; mt < 2; ++mt) {
            const unsigned short* ar = &ldsA[buf][wm + 16 * mt + lh][0];
            v8bf lo = *(const v8bf*)(ar + (half ? 16 : 0));
            v8bf hi = *(const v8bf*)(ar + (half ? 24 : 8));
            af[mt] = combine16(lo, hi);
        }
#pragma unroll
        for (int nt = 0; nt < 4; ++nt) {
            v16bf bfr = *(const v16bf*)&ldsBt[buf][wn + 16 * nt + lh][half ? 16 : 0];
#pragma unroll
            for (int mt = 0; mt < 2; ++mt)
                acc[mt][nt] = __builtin_amdgcn_wmma_f32_16x16x32_bf16(
                    false, af[mt], false, bfr, (short)0, acc[mt][nt], false, false);
        }

        async_wait0();      // next-tile async copies landed in LDS
        __syncthreads();    // (compiler also waits DScnt for the Bt scatter)
        buf ^= 1;
    }

#pragma unroll
    for (int mt = 0; mt < 2; ++mt)
#pragma unroll
        for (int nt = 0; nt < 4; ++nt)
#pragma unroll
            for (int r = 0; r < 8; ++r) {
                int row = m0 + wm + 16 * mt + r + 8 * half;
                int col = n0 + wn + 16 * nt + lh;
                float v = acc[mt][nt][r];
                if (storef32) Cf[(size_t)row * N + col] = v;
                else          Cb[(size_t)row * N + col] = f2bf(v);
            }
}

// ------------------------------------------------------------ flash attention
// qb: (B,NH,T,128) bf16 ; kb/vb: (B,NKV,T,128) bf16 ; ob: (B,T,NH*128) bf16
// grid: (B*NH, T/64), block 128 (4 waves); wave w owns q rows [q0+16w, q0+16w+16)
__global__ __launch_bounds__(128) void flash_attn_kernel(
        const unsigned short* __restrict__ qb, const unsigned short* __restrict__ kb,
        const unsigned short* __restrict__ vb, unsigned short* __restrict__ ob,
        int Bsz, int T) {
    __shared__ alignas(32) unsigned short ldsK[32][128];   // [kt][d]
    __shared__ alignas(32) unsigned short ldsVt[128][32];  // [d][kt]
    __shared__ alignas(32) unsigned short ldsP[4][16][32]; // per-wave [m][kt]

    int bh = blockIdx.x;
    int b = bh / NH, h = bh % NH, kv = h / (NH / NKV);
    int q0 = blockIdx.y * 64;
    int tid = threadIdx.x, w = tid >> 5, lane = tid & 31;
    int lh = lane & 15, half = lane >> 4;

    const unsigned short* qbase = qb + ((size_t)((size_t)b * NH + h) * T) * HD;
    const unsigned short* kbase = kb + ((size_t)((size_t)b * NKV + kv) * T) * HD;
    const unsigned short* vbase = vb + ((size_t)((size_t)b * NKV + kv) * T) * HD;

    // preload Q A-fragments (16 rows x 128 d)
    v16bf qa[4];
    {
        const unsigned short* qr = qbase + (size_t)(q0 + 16 * w + lh) * HD;
#pragma unroll
        for (int dv = 0; dv < 4; ++dv) {
            int d0 = 32 * dv + (half ? 16 : 0);
            qa[dv] = combine16(*(const v8bf*)(qr + d0), *(const v8bf*)(qr + d0 + 8));
        }
    }

    float m_r[8], l_r[8];
    v8f acc[8] = {};
#pragma unroll
    for (int r = 0; r < 8; ++r) { m_r[r] = -INFINITY; l_r[r] = 0.f; }

    const float scale = 0.08838834764831845f;  // 1/sqrt(128)
    int ktend = q0 + 64;
    if (ktend > T) ktend = T;

    for (int kt0 = 0; kt0 < ktend; kt0 += 32) {
        // stage K tile via async global->LDS; V tile transposed with DS scatter
        {
            int kt = tid >> 2, dc = (tid & 3) * 32;
            const unsigned short* ks = kbase + (size_t)(kt0 + kt) * HD + dc;
            async_ld16(&ldsK[kt][dc],      ks);
            async_ld16(&ldsK[kt][dc + 8],  ks + 8);
            async_ld16(&ldsK[kt][dc + 16], ks + 16);
            async_ld16(&ldsK[kt][dc + 24], ks + 24);
            const unsigned short* vs = vbase + (size_t)(kt0 + kt) * HD + dc;
#pragma unroll
            for (int i = 0; i < 32; ++i) ldsVt[dc + i][kt] = vs[i];
        }
        async_wait0();
        __syncthreads();

        // S = Q * K^T  (two 16x16 C-tiles, contraction d=128 in 4 WMMA steps)
        v8f s[2] = {};
#pragma unroll
        for (int j = 0; j < 2; ++j)
#pragma unroll
            for (int dv = 0; dv < 4; ++dv) {
                v16bf kf = *(const v16bf*)&ldsK[16 * j + lh][32 * dv + (half ? 16 : 0)];
                s[j] = __builtin_amdgcn_wmma_f32_16x16x32_bf16(
                    false, qa[dv], false, kf, (short)0, s[j], false, false);
            }

        // scale + causal mask (C layout: row = r+8*half, col = 16j+lh)
#pragma unroll
        for (int j = 0; j < 2; ++j)
#pragma unroll
            for (int r = 0; r < 8; ++r) {
                int rowg = q0 + 16 * w + r + 8 * half;
                int colg = kt0 + 16 * j + lh;
                float v = s[j][r] * scale;
                s[j][r] = (colg > rowg) ? -INFINITY : v;
            }

        // online softmax per row; write P (bf16) to per-wave LDS staging
#pragma unroll
        for (int r = 0; r < 8; ++r) {
            float tmax = fmaxf(s[0][r], s[1][r]);
#pragma unroll
            for (int off = 1; off < 16; off <<= 1)
                tmax = fmaxf(tmax, __shfl_xor(tmax, off, 32));
            float nm = fmaxf(m_r[r], tmax);
            float nmS = (nm < -1e30f) ? 0.f : nm;
            float p0 = __expf(s[0][r] - nmS);
            float p1 = __expf(s[1][r] - nmS);
            float alpha = __expf(fminf(m_r[r] - nmS, 0.f));
            float psum = p0 + p1;
#pragma unroll
            for (int off = 1; off < 16; off <<= 1)
                psum += __shfl_xor(psum, off, 32);
            l_r[r] = l_r[r] * alpha + psum;
            m_r[r] = nm;
#pragma unroll
            for (int nt = 0; nt < 8; ++nt) acc[nt][r] *= alpha;
            int mrow = r + 8 * half;
            ldsP[w][mrow][lh]      = f2bf(p0);
            ldsP[w][mrow][16 + lh] = f2bf(p1);
        }
        __syncthreads();  // order P stores (u16) before vector reloads

        // P (A-frag) * V^T (B-frags) accumulate into O
        int c0 = half ? 16 : 0;
        v16bf pf = combine16(*(const v8bf*)&ldsP[w][lh][c0],
                             *(const v8bf*)&ldsP[w][lh][c0 + 8]);
#pragma unroll
        for (int nt = 0; nt < 8; ++nt) {
            v16bf vf = *(const v16bf*)&ldsVt[16 * nt + lh][half ? 16 : 0];
            acc[nt] = __builtin_amdgcn_wmma_f32_16x16x32_bf16(
                false, pf, false, vf, (short)0, acc[nt], false, false);
        }
        __syncthreads();  // before next tile overwrites ldsK/ldsVt
    }

    // epilogue: O / l -> ob (B,T,NH*128) bf16
#pragma unroll
    for (int nt = 0; nt < 8; ++nt)
#pragma unroll
        for (int r = 0; r < 8; ++r) {
            int rowg = q0 + 16 * w + r + 8 * half;
            int colg = h * HD + 16 * nt + lh;
            float v = acc[nt][r] / l_r[r];
            ob[(size_t)((size_t)b * T + rowg) * (NH * HD) + colg] = f2bf(v);
        }
}

// -------------------------------------------------------------------- launch
extern "C" void kernel_launch(void* const* d_in, const int* in_sizes, int n_in,
                              void* d_out, int out_size, void* d_ws, size_t ws_size,
                              hipStream_t stream) {
    const float* x  = (const float*)d_in[0];
    const float* Wq = (const float*)d_in[1];
    const float* Wk = (const float*)d_in[2];
    const float* Wv = (const float*)d_in[3];
    const float* Wo = (const float*)d_in[4];

    const int T = SEQT;
    const int B = in_sizes[0] / (T * D_MODEL);   // = 2
    const int M = B * T;                          // 4096

    char* p = (char*)d_ws;
    auto alloc = [&](size_t bytes) -> unsigned short* {
        unsigned short* r = (unsigned short*)p;
        p += (bytes + 255) & ~(size_t)255;
        return r;
    };
    unsigned short* xb   = alloc((size_t)M * D_MODEL * 2);
    unsigned short* wqb  = alloc((size_t)D_MODEL * (NH * HD) * 2);
    unsigned short* wkb  = alloc((size_t)D_MODEL * (NKV * HD) * 2);
    unsigned short* wvb  = alloc((size_t)D_MODEL * (NKV * HD) * 2);
    unsigned short* wob  = alloc((size_t)(NH * HD) * D_MODEL * 2);
    unsigned short* qtmp = alloc((size_t)M * (NH * HD) * 2);   // reused as ob
    unsigned short* ktmp = alloc((size_t)M * (NKV * HD) * 2);
    unsigned short* vtmp = alloc((size_t)M * (NKV * HD) * 2);
    unsigned short* qb   = alloc((size_t)M * (NH * HD) * 2);
    unsigned short* kb   = alloc((size_t)M * (NKV * HD) * 2);
    unsigned short* vb   = alloc((size_t)M * (NKV * HD) * 2);
    unsigned short* ob   = qtmp;  // qtmp is dead after rope_permute

    auto conv = [&](const float* src, unsigned short* dst, size_t n) {
        int blocks = (int)((n + 255) / 256);
        if (blocks > 65535) blocks = 65535;
        f32_to_bf16_kernel<<<blocks, 256, 0, stream>>>(src, dst, n);
    };
    conv(x,  xb,  (size_t)M * D_MODEL);
    conv(Wq, wqb, (size_t)D_MODEL * NH * HD);
    conv(Wk, wkb, (size_t)D_MODEL * NKV * HD);
    conv(Wv, wvb, (size_t)D_MODEL * NKV * HD);
    conv(Wo, wob, (size_t)(NH * HD) * D_MODEL);

    // QKV projections (bf16 WMMA GEMM)
    gemm_bf16_wmma<<<dim3((NH * HD) / 128, M / 128), 256, 0, stream>>>(
        xb, wqb, nullptr, qtmp, M, NH * HD, D_MODEL, 0);
    gemm_bf16_wmma<<<dim3((NKV * HD) / 128, M / 128), 256, 0, stream>>>(
        xb, wkb, nullptr, ktmp, M, NKV * HD, D_MODEL, 0);
    gemm_bf16_wmma<<<dim3((NKV * HD) / 128, M / 128), 256, 0, stream>>>(
        xb, wvb, nullptr, vtmp, M, NKV * HD, D_MODEL, 0);

    // RoPE + head-major permute
    rope_permute_kernel<<<dim3(B * NH, T), 64, 0, stream>>>(qtmp, qb, NH, T, 1);
    rope_permute_kernel<<<dim3(B * NKV, T), 64, 0, stream>>>(ktmp, kb, NKV, T, 1);
    rope_permute_kernel<<<dim3(B * NKV, T), 64, 0, stream>>>(vtmp, vb, NKV, T, 0);

    // flash attention
    flash_attn_kernel<<<dim3(B * NH, T / 64), 128, 0, stream>>>(qb, kb, vb, ob, B, T);

    // output projection -> fp32 d_out
    gemm_bf16_wmma<<<dim3(D_MODEL / 128, M / 128), 256, 0, stream>>>(
        ob, wob, (float*)d_out, nullptr, M, D_MODEL, NH * HD, 1);
}
